// depth_reg_54700703482077
// MI455X (gfx1250) — matrix operand order repaired and checked by reading, not compiled, not run on hardware
//
#include <hip/hip_runtime.h>

typedef __attribute__((ext_vector_type(16))) _Float16 v16h;
typedef __attribute__((ext_vector_type(8)))  _Float16 v8h;
typedef __attribute__((ext_vector_type(8)))  float    v8f;

static constexpr int N0 = 8000;    // input points
static constexpr int C0 = 128;
static constexpr int N1 = 64000;   // 8*N0
static constexpr int N2 = 512000;  // 8*N1

// ---------------------------------------------------------------------------
// WMMA fragment helpers (CDNA5 16x16x32 f16, wave32 layouts per CDNA5 ISA)
// A (16x32, row-major [M,K]): lane m=l&15; lanes0-15 K-chunks {0..7,16..23},
//   lanes16-31 K-chunks {8..15,24..31} -> two contiguous 16B loads.
// B (32x16, stored transposed [N,K]): lane n=l&15; lanes0-15 K-chunks {0..7,16..23},
//   lanes16-31 K-chunks {8..15,24..31} -> two contiguous 16B loads.
// D (16x16 f32): VGPR r, lane l -> row r + (l>=16 ? 8:0), col l&15.
// ---------------------------------------------------------------------------
static __device__ __forceinline__ v16h cat8(v8h lo, v8h hi) {
  return __builtin_shufflevector(lo, hi, 0, 1, 2, 3, 4, 5, 6, 7,
                                 8, 9, 10, 11, 12, 13, 14, 15);
}

static __device__ __forceinline__ v16h wmma_load_a(const _Float16* __restrict__ A, int lda,
                                                   int row_base, int k0, int lane) {
  const int m    = lane & 15;
  const int koff = (lane & 16) ? 8 : 0;
  const _Float16* p = A + (size_t)(row_base + m) * (size_t)lda + (size_t)(k0 + koff);
  return cat8(*(const v8h*)p, *(const v8h*)(p + 16));
}

static __device__ __forceinline__ v16h wmma_load_b(const _Float16* __restrict__ BT, int ldb,
                                                   int col_base, int k0, int lane) {
  const int n    = lane & 15;
  const int koff = (lane & 16) ? 8 : 0;
  const _Float16* p = BT + (size_t)(col_base + n) * (size_t)ldb + (size_t)(k0 + koff);
  return cat8(*(const v8h*)p, *(const v8h*)(p + 16));  // 16B aligned: ldb,k0 multiples of 16 elems, koff 0/8
}

static __device__ __forceinline__ void wmma_store_d(float* __restrict__ C, int ldc, int row_base,
                                                    int col_base, int lane, v8f acc) {
  const int n  = col_base + (lane & 15);
  const int mo = (lane & 16) ? 8 : 0;
#pragma unroll
  for (int r = 0; r < 8; ++r)
    C[(size_t)(row_base + r + mo) * (size_t)ldc + n] = acc[r];
}

// ---------------------------------------------------------------------------
// Dense GEMM: C[M,N] (f32) = A[M,KK] (f16) * BT[N,KK]^T (f16).
// One wave = 16 rows x NT*16 cols. KK, NT compile-time; N == ngroups*NT*16
// exactly, so the MAC loop has no bounds checks and control flow is scalar.
// ---------------------------------------------------------------------------
template <int KK, int NT>
__global__ __launch_bounds__(256) void k_gemm_wmma(const _Float16* __restrict__ A,
                                                   const _Float16* __restrict__ BT,
                                                   float* __restrict__ C,
                                                   int M, int N, int ngroups) {
  const int lane = (int)(threadIdx.x & 31);
  const int wib  = __builtin_amdgcn_readfirstlane((int)(threadIdx.x >> 5));
  const int wave = (int)blockIdx.x * (int)(blockDim.x >> 5) + wib;   // SGPR-uniform
  const int mtile = wave / ngroups;
  const int ng    = wave - mtile * ngroups;
  if (mtile * 16 >= M) return;           // uniform scalar branch, EXEC stays all-1
  const int row = mtile * 16;
  const int c0  = ng * (NT * 16);
  v8f acc[NT];
#pragma unroll
  for (int t = 0; t < NT; ++t) acc[t] = v8f{};
  for (int k0 = 0; k0 < KK; k0 += 32) {
    v16h a = wmma_load_a(A, KK, row, k0, lane);
#pragma unroll
    for (int t = 0; t < NT; ++t) {
      v16h b = wmma_load_b(BT, KK, c0 + t * 16, k0, lane);
      acc[t] = __builtin_amdgcn_wmma_f32_16x16x32_f16(false, a, false, b,
                                                      (short)0, acc[t], false, false);
    }
  }
#pragma unroll
  for (int t = 0; t < NT; ++t) wmma_store_d(C, N, row, c0 + t * 16, lane, acc[t]);
}

// ---------------------------------------------------------------------------
// Sparse 3^3 conv: out[M,CT*16] = sum_{ko<27} (A[idx[ko,m]] * mask[ko,m]) @ W[ko]
// Gathered/masked A fragments; per-ko B slices stay hot in L2.
// ---------------------------------------------------------------------------
template <int KK, int CT>
__global__ __launch_bounds__(256) void k_conv3_wmma(const _Float16* __restrict__ A,
                                                    const int* __restrict__ idx,
                                                    const float* __restrict__ mask,
                                                    const _Float16* __restrict__ BT,
                                                    float* __restrict__ C, int M) {
  const int lane = (int)(threadIdx.x & 31);
  const int wib  = __builtin_amdgcn_readfirstlane((int)(threadIdx.x >> 5));
  const int wave = (int)blockIdx.x * (int)(blockDim.x >> 5) + wib;   // SGPR-uniform
  if (wave * 16 >= M) return;            // uniform
  const int m    = wave * 16 + (lane & 15);
  const int koff = (lane & 16) ? 8 : 0;
  v8f acc[CT];
#pragma unroll
  for (int t = 0; t < CT; ++t) acc[t] = v8f{};
  for (int ko = 0; ko < 27; ++ko) {
    const int g = idx[(size_t)ko * M + m];
    const _Float16 hm = (_Float16)mask[(size_t)ko * M + m];
    v16h hv;
#pragma unroll
    for (int i = 0; i < 16; ++i) hv[i] = hm;   // splat
    const _Float16* __restrict__ BTk  = BT + (size_t)ko * (size_t)(CT * 16) * (size_t)KK;
    const _Float16* __restrict__ arow = A + (size_t)g * (size_t)KK + koff;
#pragma unroll
    for (int k0 = 0; k0 < KK; k0 += 32) {
      const _Float16* p = arow + k0;
      v16h a = cat8(*(const v8h*)p, *(const v8h*)(p + 16)) * hv;  // mask fold (v_pk_mul_f16)
#pragma unroll
      for (int t = 0; t < CT; ++t) {
        v16h b = wmma_load_b(BTk, KK, t * 16, k0, lane);
        acc[t] = __builtin_amdgcn_wmma_f32_16x16x32_f16(false, a, false, b,
                                                        (short)0, acc[t], false, false);
      }
    }
  }
#pragma unroll
  for (int t = 0; t < CT; ++t) wmma_store_d(C, CT * 16, wave * 16, t * 16, lane, acc[t]);
}

// ---------------------------------------------------------------------------
// Small utility kernels
// ---------------------------------------------------------------------------
__global__ void k_zero(float* p, int n) {
  int i = blockIdx.x * blockDim.x + threadIdx.x;
  if (i < n) p[i] = 0.f;
}

__global__ void k_f32_to_f16(const float* __restrict__ x, _Float16* __restrict__ y, int n) {
  int i = blockIdx.x * blockDim.x + threadIdx.x;
  if (i < n) y[i] = (_Float16)x[i];
}

// w [Kk,Cin,Cout] f32 -> out f16 with out[(k*Cout+d)*Cin + c] = w[(k*Cin+c)*Cout + d]
__global__ void k_wconvert(const float* __restrict__ w, _Float16* __restrict__ out,
                           int Kk, int Cin, int Cout) {
  int i = blockIdx.x * blockDim.x + threadIdx.x;
  int total = Kk * Cin * Cout;
  if (i >= total) return;
  int c = i % Cin;
  int r = i / Cin;
  int d = r % Cout;
  int k = r / Cout;
  out[i] = (_Float16)w[((size_t)k * Cin + c) * (size_t)Cout + d];
}

// Per-channel sum & sumsq. blockDim.x == 192 (divisible by C in {32,64,96});
// each thread owns one channel, privately reduces strided rows, one LDS
// combine, then 2 global atomics per channel per block.
__global__ __launch_bounds__(192) void k_stats(const float* __restrict__ x,
                                               float* __restrict__ sums,
                                               float* __restrict__ sumsq,
                                               int rows, int C) {
  __shared__ float smA[192], smB[192];
  const int t    = (int)threadIdx.x;
  const int c    = t % C;
  const int rloc = t / C;
  const int rpi  = (int)blockDim.x / C;     // rows per block-iteration
  float s = 0.f, q = 0.f;
  for (int r = (int)blockIdx.x * rpi + rloc; r < rows; r += (int)gridDim.x * rpi) {
    float v = x[(size_t)r * (size_t)C + c];
    s += v;
    q += v * v;
  }
  smA[t] = s;
  smB[t] = q;
  __syncthreads();
  if (t < C) {
    for (int j = 1; j < rpi; ++j) { s += smA[t + j * C]; q += smB[t + j * C]; }
    atomicAdd(&sums[c], s);
    atomicAdd(&sumsq[c], q);
  }
}

__global__ void k_bn_finalize(const float* sums, const float* sumsq, const float* g,
                              const float* b, float* sc, float* sh, float invM, int C) {
  int c = blockIdx.x * blockDim.x + threadIdx.x;
  if (c >= C) return;
  float m   = sums[c] * invM;
  float var = fmaf(-m, m, sumsq[c] * invM);
  float s   = g[c] * rsqrtf(var + 1e-5f);
  sc[c] = s;
  sh[c] = fmaf(-m, s, b[c]);
}

__global__ void k_bn_relu_to_f16(const float* __restrict__ x, const float* __restrict__ sc,
                                 const float* __restrict__ sh, _Float16* __restrict__ y,
                                 int total, int C) {
  int i = blockIdx.x * blockDim.x + threadIdx.x;
  if (i >= total) return;
  int c = i % C;
  y[i] = (_Float16)fmaxf(fmaf(sc[c], x[i], sh[c]), 0.f);
}

// stage-4 apply: BN+ReLU -> f16 activations AND feature-mean -> intensity (C=32)
__global__ void k_bn_relu_h4(const float* __restrict__ x, const float* __restrict__ sc,
                             const float* __restrict__ sh, _Float16* __restrict__ y,
                             float* __restrict__ inten, int M) {
  int r = blockIdx.x * blockDim.x + threadIdx.x;
  if (r >= M) return;
  const float* xr = x + (size_t)r * 32;
  _Float16* yr = y + (size_t)r * 32;
  float s = 0.f;
#pragma unroll
  for (int c = 0; c < 32; ++c) {
    float v = fmaxf(fmaf(sc[c], xr[c], sh[c]), 0.f);
    yr[c] = (_Float16)v;
    s += v;
  }
  inten[r] = s * (1.f / 32.f);
}

__global__ void k_bn_relu_inplace(float* __restrict__ x, const float* __restrict__ sc,
                                  const float* __restrict__ sh, int total, int C) {
  int i = blockIdx.x * blockDim.x + threadIdx.x;
  if (i >= total) return;
  int c = i % C;
  x[i] = fmaxf(fmaf(sc[c], x[i], sh[c]), 0.f);
}

// ---------------------------------------------------------------------------
static inline int cdiv(int a, int b) { return (a + b - 1) / b; }

extern "C" void kernel_launch(void* const* d_in, const int* in_sizes, int n_in,
                              void* d_out, int out_size, void* d_ws, size_t ws_size,
                              hipStream_t stream) {
  (void)in_sizes; (void)n_in; (void)out_size; (void)ws_size;

  const float* x     = (const float*)d_in[0];
  const float* w_tr1 = (const float*)d_in[1];
  const float* g1    = (const float*)d_in[2];
  const float* b1    = (const float*)d_in[3];
  const int*   idx1  = (const int*)  d_in[4];
  const float* mask1 = (const float*)d_in[5];
  const float* w_c1  = (const float*)d_in[6];
  const float* g2    = (const float*)d_in[7];
  const float* b2    = (const float*)d_in[8];
  const float* w_tr2 = (const float*)d_in[9];
  const float* g3    = (const float*)d_in[10];
  const float* b3    = (const float*)d_in[11];
  const int*   idx2  = (const int*)  d_in[12];
  const float* mask2 = (const float*)d_in[13];
  const float* w_c2  = (const float*)d_in[14];
  const float* g4    = (const float*)d_in[15];
  const float* b4    = (const float*)d_in[16];
  const float* w_dec = (const float*)d_in[17];
  const float* g5    = (const float*)d_in[18];
  const float* b5    = (const float*)d_in[19];

  float* intensity = (float*)d_out;          // [N2]
  float* remi      = (float*)d_out + N2;     // [N2, 96] raw GEMM output -> BN in place

  // ---- workspace arena (all sizes already multiples of 256B) ----
  char* ws = (char*)d_ws;
  size_t off = 0;
  auto take = [&](size_t bytes) -> char* {
    char* p = ws + off;
    off += (bytes + 255) & ~(size_t)255;
    return p;
  };
  _Float16* xb   = (_Float16*)take((size_t)N0 * C0 * 2);        // x in f16
  _Float16* w1t  = (_Float16*)take((size_t)768 * 128 * 2);      // stage1 weights [768][128]
  _Float16* wc1t = (_Float16*)take((size_t)27 * 64 * 96 * 2);   // conv1 weights 27x[64][96]
  _Float16* w3t  = (_Float16*)take((size_t)512 * 64 * 2);       // stage3 weights [512][64]
  _Float16* wc2t = (_Float16*)take((size_t)27 * 32 * 64 * 2);   // conv2 weights 27x[32][64]
  _Float16* wdt  = (_Float16*)take((size_t)96 * 32 * 2);        // decode weights [96][32]
  float* stats   = (float*)take(4 * 768 * 4);
  float* sums = stats, *sumsq = stats + 768, *scl = stats + 1536, *shf = stats + 2304;
  char* bufA = take((size_t)N2 * 64 * 4);   // r1 [N1,96] f32 | r3 [N2,64] f32
  char* bufB = take((size_t)N2 * 32 * 4);   // h1 f16 [N1,96] | r4 [N2,32] f32
  char* bufC = take((size_t)N2 * 64 * 2);   // r2 [N1,64] f32 | h3 f16 [N2,64]
  char* bufD = take((size_t)N2 * 32 * 2);   // h2 f16 [N1,64] | h4 f16 [N2,32]

  float*    r1  = (float*)bufA;
  _Float16* h1b = (_Float16*)bufB;
  float*    r2  = (float*)bufC;
  _Float16* h2b = (_Float16*)bufD;
  float*    r3  = (float*)bufA;
  _Float16* h3b = (_Float16*)bufC;
  float*    r4  = (float*)bufB;
  _Float16* h4b = (_Float16*)bufD;

  const int TPB = 256;  // 8 waves per block on wave32

  auto run_bn_stats = [&](const float* raw, const float* g, const float* b, int rows, int C) {
    k_zero<<<cdiv(1536, TPB), TPB, 0, stream>>>(sums, 1536);
    k_stats<<<768, 192, 0, stream>>>(raw, sums, sumsq, rows, C);
    k_bn_finalize<<<1, TPB, 0, stream>>>(sums, sumsq, g, b, scl, shf, 1.f / (float)rows, C);
  };

  // ---- precision conversion / weight repack ----
  k_f32_to_f16<<<cdiv(N0 * C0, TPB), TPB, 0, stream>>>(x, xb, N0 * C0);
  k_wconvert<<<cdiv(8 * 128 * 96, TPB), TPB, 0, stream>>>(w_tr1, w1t, 8, 128, 96);
  k_wconvert<<<cdiv(27 * 96 * 64, TPB), TPB, 0, stream>>>(w_c1, wc1t, 27, 96, 64);
  k_wconvert<<<cdiv(8 * 64 * 64, TPB), TPB, 0, stream>>>(w_tr2, w3t, 8, 64, 64);
  k_wconvert<<<cdiv(27 * 64 * 32, TPB), TPB, 0, stream>>>(w_c2, wc2t, 27, 64, 32);
  k_wconvert<<<cdiv(32 * 96, TPB), TPB, 0, stream>>>(w_dec, wdt, 1, 32, 96);

  // ---- stage 1: convtr2 as dense GEMM [8000,128]x[128,768] -> r1 == [N1,96] ----
  k_gemm_wmma<128, 4><<<cdiv((N0 / 16) * 12, 8), TPB, 0, stream>>>(xb, w1t, r1, N0, 768, 12);
  run_bn_stats(r1, g1, b1, N1, 96);
  k_bn_relu_to_f16<<<cdiv(N1 * 96, TPB), TPB, 0, stream>>>(r1, scl, shf, h1b, N1 * 96, 96);

  // ---- stage 2: sparse conv3 [N1,96] -> r2 [N1,64] ----
  k_conv3_wmma<96, 4><<<cdiv(N1 / 16, 8), TPB, 0, stream>>>(h1b, idx1, mask1, wc1t, r2, N1);
  run_bn_stats(r2, g2, b2, N1, 64);
  k_bn_relu_to_f16<<<cdiv(N1 * 64, TPB), TPB, 0, stream>>>(r2, scl, shf, h2b, N1 * 64, 64);

  // ---- stage 3: convtr2 as dense GEMM [N1,64]x[64,512] -> r3 == [N2,64] ----
  k_gemm_wmma<64, 4><<<cdiv((N1 / 16) * 8, 8), TPB, 0, stream>>>(h2b, w3t, r3, N1, 512, 8);
  run_bn_stats(r3, g3, b3, N2, 64);
  k_bn_relu_to_f16<<<cdiv(N2 * 64, TPB), TPB, 0, stream>>>(r3, scl, shf, h3b, N2 * 64, 64);

  // ---- stage 4: sparse conv3 [N2,64] -> r4 [N2,32] ----
  k_conv3_wmma<64, 2><<<cdiv(N2 / 16, 8), TPB, 0, stream>>>(h3b, idx2, mask2, wc2t, r4, N2);
  run_bn_stats(r4, g4, b4, N2, 32);
  k_bn_relu_h4<<<cdiv(N2, TPB), TPB, 0, stream>>>(r4, scl, shf, h4b, intensity, N2);

  // ---- stage 5: decode GEMM [N2,32]x[32,96] -> d_out, BN+ReLU in place ----
  k_gemm_wmma<32, 6><<<cdiv((N2 / 16) * 1, 8), TPB, 0, stream>>>(h4b, wdt, remi, N2, 96, 1);
  run_bn_stats(remi, g5, b5, N2, 96);
  k_bn_relu_inplace<<<cdiv(N2 * 96, TPB), TPB, 0, stream>>>(remi, scl, shf, N2 * 96, 96);
}
